// EMAU_38817914422016
// MI455X (gfx1250) — matrix operand order repaired and checked by reading, not compiled, not run on hardware
//
#include <hip/hip_runtime.h>

#define Cc 512
#define Kk 64
#define Bb 16
#define Nn 4096   // H*W = 64*64

typedef __attribute__((ext_vector_type(16))) __bf16 v16bf;
typedef __attribute__((ext_vector_type(8)))  float  v8f;

union FragU { v16bf v; uint4 q[2]; };

__device__ __forceinline__ unsigned short f2bf(float f) {
  union { float f; unsigned u; } v; v.f = f;
  unsigned u = v.u;
  unsigned r = u + 0x7FFFu + ((u >> 16) & 1u);  // round-to-nearest-even
  return (unsigned short)(r >> 16);
}

// A fragment: 16x32 bf16 tile, rows striped over lanes, K over VGPRs.
// lane (m = lane&15, h = lane>>4): elems 0..7 -> K = kb + h*8 + (0..7)
//                                  elems 8..15 -> K = kb + 16 + h*8 + (0..7)
__device__ __forceinline__ FragU load_frag_a(const unsigned short* A, int lda,
                                             int row, int kb, int lane) {
  int m = lane & 15, h = lane >> 4;
  const unsigned short* p = A + (size_t)(row + m) * lda + kb + h * 8;
  FragU f;
  f.q[0] = *(const uint4*)p;
  f.q[1] = *(const uint4*)(p + 16);
  return f;
}

// B fragment from Bt (row-major N x K, i.e. B column-major):
// lane (n = lane&15, h = lane>>4): elems 0..7 -> K = kb + h*8 + (0..7), elems 8..15 -> K = kb + 16 + h*8 + (0..7)
__device__ __forceinline__ FragU load_frag_b(const unsigned short* Bt, int ldb,
                                             int col, int kb, int lane) {
  int n = lane & 15, h = lane >> 4;
  const unsigned short* p = Bt + (size_t)(col + n) * ldb + kb + h * 8;
  FragU f;
  f.q[0] = *(const uint4*)p;
  f.q[1] = *(const uint4*)(p + 16);
  return f;
}

// NT GEMM: C[m,n] = sum_k A[m*lda+k] * Bt[n*ldb+k], bf16 in, fp32 accumulate.
// Block: 8 waves = 256 threads arranged 4(M) x 2(N).
// Wave tile: (WMT*16) x (WNT*16)  ->  block tile (64*WMT) x (32*WNT).
// EPI: 0 = fp32 store; 1 = +bias, bf16 store + bf16 transposed store;
//      2 = relu, bf16 transposed store only; 3 = fp32 store + channel stats.
template<int EPI, int WMT, int WNT>
__global__ void __launch_bounds__(256)
gemm_nt(const unsigned short* __restrict__ A, const unsigned short* __restrict__ Bt,
        int M, int Kd, int lda, int ldb, int ldc,
        size_t sA, size_t sB, size_t sO0, size_t sO1,
        const float* __restrict__ bias,
        float* __restrict__ outF,
        unsigned short* __restrict__ outB,
        unsigned short* __restrict__ outT,
        float* __restrict__ chsum, float* __restrict__ chsq)
{
  int b = blockIdx.z;
  A  += (size_t)b * sA;
  Bt += (size_t)b * sB;
  int lane = threadIdx.x & 31;
  int wave = threadIdx.x >> 5;
  int mW = blockIdx.y * (64 * WMT) + (wave >> 1) * (WMT * 16);
  int nW = blockIdx.x * (32 * WNT) + (wave & 1)  * (WNT * 16);

  v8f zeroV = {0.f,0.f,0.f,0.f,0.f,0.f,0.f,0.f};
  v8f acc[WMT][WNT];
  #pragma unroll
  for (int i = 0; i < WMT; i++)
    #pragma unroll
    for (int j = 0; j < WNT; j++) acc[i][j] = zeroV;

  for (int kk = 0; kk < Kd; kk += 32) {
    FragU af[WMT], bg[WNT];
    #pragma unroll
    for (int i = 0; i < WMT; i++) af[i] = load_frag_a(A, lda, mW + i * 16, kk, lane);
    #pragma unroll
    for (int j = 0; j < WNT; j++) bg[j] = load_frag_b(Bt, ldb, nW + j * 16, kk, lane);
    #pragma unroll
    for (int i = 0; i < WMT; i++) {
      #pragma unroll
      for (int j = 0; j < WNT; j++) {
        acc[i][j] = __builtin_amdgcn_wmma_f32_16x16x32_bf16(
            false, af[i].v, false, bg[j].v, (short)0, acc[i][j], false, false);
      }
    }
  }

  // D layout: VGPR r holds row m = base + (lane>=16 ? 8 : 0) + r, col = lane&15
  int cil = lane & 15;
  int h8  = (lane >> 4) * 8;
  #pragma unroll
  for (int fm = 0; fm < WMT; fm++) {
    #pragma unroll
    for (int fn = 0; fn < WNT; fn++) {
      #pragma unroll
      for (int r = 0; r < 8; r++) {
        int m = mW + fm * 16 + h8 + r;
        int n = nW + fn * 16 + cil;
        float v = acc[fm][fn][r];
        if (EPI == 0) {
          outF[(size_t)b * sO0 + (size_t)m * ldc + n] = v;
        } else if (EPI == 1) {
          v += bias[m];
          unsigned short bf = f2bf(v);
          outB[(size_t)b * sO0 + (size_t)m * ldc + n] = bf;
          outT[(size_t)b * sO1 + (size_t)n * M + m]   = bf;
        } else if (EPI == 2) {
          v = v > 0.f ? v : 0.f;
          outT[(size_t)b * sO1 + (size_t)n * M + m] = f2bf(v);
        } else {
          outF[(size_t)b * sO0 + (size_t)m * ldc + n] = v;
          atomicAdd(&chsum[m], v);
          atomicAdd(&chsq[m],  v * v);
        }
      }
    }
  }
}

__global__ void k_zero(float* p, int n) {
  int i = blockIdx.x * 256 + threadIdx.x;
  if (i < n) p[i] = 0.f;
}

__global__ void k_cast_bf(const float* __restrict__ in, unsigned short* __restrict__ out, int n) {
  int i = blockIdx.x * 256 + threadIdx.x;
  if (i < n) out[i] = f2bf(in[i]);
}

// x [b][c][n] fp32 -> xT [b][n][c] bf16
__global__ void __launch_bounds__(256) k_xT(const float* __restrict__ x,
                                            unsigned short* __restrict__ xT) {
  int i = blockIdx.x * 256 + threadIdx.x;           // over B*C*N = 2^25
  int b = i >> 21;
  int c = (i >> 12) & (Cc - 1);
  int n = i & (Nn - 1);
  xT[(((size_t)b << 12) + n) * Cc + c] = f2bf(x[i]);
}

// mu (1,C,K) fp32 -> per-batch muT [b][k][c] bf16
__global__ void k_mu_init(const float* __restrict__ mu, unsigned short* __restrict__ muT) {
  int i = blockIdx.x * 256 + threadIdx.x;           // over B*C*K
  int b = i >> 15;
  int c = (i >> 6) & (Cc - 1);
  int k = i & (Kk - 1);
  muT[((size_t)(b * Kk + k)) * Cc + c] = f2bf(mu[c * Kk + k]);
}

// softmax over k (in place on z [b][n][k]) + column sums over n -> colsum [b][k]
__global__ void __launch_bounds__(256) k_softmax(float* __restrict__ z,
                                                 float* __restrict__ colsum) {
  __shared__ float ls[Kk];
  int idx = blockIdx.x * 256 + threadIdx.x;         // b*Nn + n; block stays in one b
  int b = idx >> 12;
  float* row = z + (size_t)idx * Kk;
  float v[Kk];
  float mx = -3.4e38f;
  #pragma unroll
  for (int k = 0; k < Kk; k++) { v[k] = row[k]; mx = fmaxf(mx, v[k]); }
  float s = 0.f;
  #pragma unroll
  for (int k = 0; k < Kk; k++) { v[k] = __expf(v[k] - mx); s += v[k]; }
  float inv = 1.f / s;
  if (threadIdx.x < Kk) ls[threadIdx.x] = 0.f;
  __syncthreads();
  #pragma unroll
  for (int k = 0; k < Kk; k++) {
    float sv = v[k] * inv;
    row[k] = sv;
    atomicAdd(&ls[k], sv);
  }
  __syncthreads();
  if (threadIdx.x < Kk) atomicAdd(&colsum[b * Kk + threadIdx.x], ls[threadIdx.x]);
}

// znk [b][n][k] bf16 = softmax z ; ztk [b][k][n] bf16 = z / (1e-6 + colsum)
__global__ void __launch_bounds__(256) k_scale(const float* __restrict__ z,
                                               const float* __restrict__ colsum,
                                               unsigned short* __restrict__ znk,
                                               unsigned short* __restrict__ ztk) {
  int idx = blockIdx.x * 256 + threadIdx.x;         // b*Nn + n
  int b = idx >> 12;
  int n = idx & (Nn - 1);
  const float* row = z + (size_t)idx * Kk;
  #pragma unroll
  for (int k = 0; k < Kk; k++) {
    float sv = row[k];
    znk[(size_t)idx * Kk + k] = f2bf(sv);
    float zn = sv / (1e-6f + colsum[b * Kk + k]);
    ztk[(((size_t)(b * Kk + k)) << 12) + n] = f2bf(zn);
  }
}

// l2 normalize mu_acc [b][c][k] over c -> muT [b][k][c] bf16 and mub [b][c][k] bf16
__global__ void __launch_bounds__(256) k_l2norm(const float* __restrict__ macc,
                                                unsigned short* __restrict__ muT,
                                                unsigned short* __restrict__ mub) {
  __shared__ float red[256];
  int b = blockIdx.x >> 6;
  int k = blockIdx.x & (Kk - 1);
  const float* src = macc + ((size_t)b * Cc) * Kk + k;
  int t = threadIdx.x;
  float v0 = src[(size_t)t * Kk];
  float v1 = src[(size_t)(t + 256) * Kk];
  red[t] = v0 * v0 + v1 * v1;
  __syncthreads();
  for (int s = 128; s > 0; s >>= 1) {
    if (t < s) red[t] += red[t + s];
    __syncthreads();
  }
  float scale = 1.f / (1e-6f + sqrtf(red[0]));
  unsigned short o0 = f2bf(v0 * scale);
  unsigned short o1 = f2bf(v1 * scale);
  muT[((size_t)(b * Kk + k)) * Cc + t]           = o0;
  muT[((size_t)(b * Kk + k)) * Cc + t + 256]     = o1;
  mub[((size_t)(b * Cc + t))       * Kk + k]     = o0;
  mub[((size_t)(b * Cc + t + 256)) * Kk + k]     = o1;
}

__global__ void k_bnstat(const float* __restrict__ chsum, const float* __restrict__ chsq,
                         float* __restrict__ meanv, float* __restrict__ rstdv) {
  int c = blockIdx.x * 256 + threadIdx.x;
  if (c < Cc) {
    const float inv = 1.f / (float)(Bb * Nn);
    float m   = chsum[c] * inv;
    float var = chsq[c] * inv - m * m;
    meanv[c] = m;
    rstdv[c] = rsqrtf(var + 1e-5f);
  }
}

__global__ void __launch_bounds__(256) k_final(const float* __restrict__ y,
                                               const float* __restrict__ x,
                                               const float* __restrict__ meanv,
                                               const float* __restrict__ rstdv,
                                               const float* __restrict__ bnw,
                                               const float* __restrict__ bnb,
                                               float* __restrict__ out) {
  int i = blockIdx.x * 256 + threadIdx.x;           // over B*C*N
  int c = (i >> 12) & (Cc - 1);
  float v = (y[i] - meanv[c]) * rstdv[c] * bnw[c] + bnb[c] + x[i];
  out[i] = v > 0.f ? v : 0.f;
}

extern "C" void kernel_launch(void* const* d_in, const int* in_sizes, int n_in,
                              void* d_out, int out_size, void* d_ws, size_t ws_size,
                              hipStream_t stream) {
  const float* x   = (const float*)d_in[0];
  const float* w1  = (const float*)d_in[1];
  const float* b1  = (const float*)d_in[2];
  const float* w2  = (const float*)d_in[3];
  const float* bnw = (const float*)d_in[4];
  const float* bnb = (const float*)d_in[5];
  const float* mu  = (const float*)d_in[6];
  float* out = (float*)d_out;

  char* ws = (char*)d_ws;
  size_t off = 0;
  auto alloc = [&](size_t bytes) -> char* {
    char* p = ws + off;
    off += (bytes + 255) & ~(size_t)255;
    return p;
  };

  unsigned short* w1b  = (unsigned short*)alloc((size_t)Cc * Cc * 2);
  unsigned short* w2b  = (unsigned short*)alloc((size_t)Cc * Cc * 2);
  unsigned short* xT   = (unsigned short*)alloc((size_t)Bb * Nn * Cc * 2);
  unsigned short* xf   = (unsigned short*)alloc((size_t)Bb * Cc * Nn * 2);
  unsigned short* xfT  = (unsigned short*)alloc((size_t)Bb * Nn * Cc * 2);
  unsigned short* muT  = (unsigned short*)alloc((size_t)Bb * Kk * Cc * 2);
  unsigned short* mub  = (unsigned short*)alloc((size_t)Bb * Cc * Kk * 2);
  float*          zf   = (float*)alloc((size_t)Bb * Nn * Kk * 4);
  float*        colsum = (float*)alloc((size_t)Bb * Kk * 4);
  unsigned short* znk  = (unsigned short*)alloc((size_t)Bb * Nn * Kk * 2);
  unsigned short* ztk  = (unsigned short*)alloc((size_t)Bb * Kk * Nn * 2);
  float*          macc = (float*)alloc((size_t)Bb * Cc * Kk * 4);
  unsigned short* xrT  = (unsigned short*)alloc((size_t)Bb * Nn * Cc * 2);
  float*          yb   = (float*)alloc((size_t)Bb * Cc * Nn * 4);
  float*         chsum = (float*)alloc(Cc * 4);
  float*         chsq  = (float*)alloc(Cc * 4);
  float*         meanv = (float*)alloc(Cc * 4);
  float*         rstdv = (float*)alloc(Cc * 4);

  // --- prep: bf16 casts + transposes ---
  k_cast_bf<<<(Cc * Cc + 255) / 256, 256, 0, stream>>>(w1, w1b, Cc * Cc);
  k_cast_bf<<<(Cc * Cc + 255) / 256, 256, 0, stream>>>(w2, w2b, Cc * Cc);
  k_xT<<<(Bb * Cc * Nn) / 256, 256, 0, stream>>>(x, xT);
  k_mu_init<<<(Bb * Cc * Kk) / 256, 256, 0, stream>>>(mu, muT);

  // --- conv1: xf[c][n] = w1 * x, + bias; also write xfT ---
  // 64x64 wave tile, block tile 256x128
  gemm_nt<1, 4, 4><<<dim3(Nn / 128, Cc / 256, Bb), 256, 0, stream>>>(
      w1b, xT, Cc, Cc, Cc, Cc, Nn,
      0, (size_t)Nn * Cc, (size_t)Cc * Nn, (size_t)Nn * Cc,
      b1, nullptr, xf, xfT, nullptr, nullptr);

  // --- EM iterations ---
  for (int it = 0; it < 3; ++it) {
    k_zero<<<(Bb * Kk + 255) / 256, 256, 0, stream>>>(colsum, Bb * Kk);
    // z[n][k] = xfT * muT^T  (Kd = C); 32x32 wave tile, block 128x64
    gemm_nt<0, 2, 2><<<dim3(Kk / 64, Nn / 128, Bb), 256, 0, stream>>>(
        xfT, muT, Nn, Cc, Cc, Cc, Kk,
        (size_t)Nn * Cc, (size_t)Kk * Cc, (size_t)Nn * Kk, 0,
        nullptr, zf, nullptr, nullptr, nullptr, nullptr);
    k_softmax<<<(Bb * Nn) / 256, 256, 0, stream>>>(zf, colsum);
    k_scale<<<(Bb * Nn) / 256, 256, 0, stream>>>(zf, colsum, znk, ztk);
    // mu_acc[c][k] = xf * ztk^T  (Kd = N); block 128x64
    gemm_nt<0, 2, 2><<<dim3(Kk / 64, Cc / 128, Bb), 256, 0, stream>>>(
        xf, ztk, Cc, Nn, Nn, Nn, Kk,
        (size_t)Cc * Nn, (size_t)Kk * Nn, (size_t)Cc * Kk, 0,
        nullptr, macc, nullptr, nullptr, nullptr, nullptr);
    k_l2norm<<<Bb * Kk, 256, 0, stream>>>(macc, muT, mub);
  }

  // --- reconstruction: xrT[n][c] = relu(mu * z^T)^T  (Kd = K); block 256x128 ---
  gemm_nt<2, 4, 4><<<dim3(Nn / 128, Cc / 256, Bb), 256, 0, stream>>>(
      mub, znk, Cc, Kk, Kk, Kk, 0,
      (size_t)Cc * Kk, (size_t)Nn * Kk, 0, (size_t)Nn * Cc,
      nullptr, nullptr, nullptr, xrT, nullptr, nullptr);

  // --- conv2 + BN stats; block 256x128 ---
  k_zero<<<(Cc + 255) / 256, 256, 0, stream>>>(chsum, Cc);
  k_zero<<<(Cc + 255) / 256, 256, 0, stream>>>(chsq, Cc);
  gemm_nt<3, 4, 4><<<dim3(Nn / 128, Cc / 256, Bb), 256, 0, stream>>>(
      w2b, xrT, Cc, Cc, Cc, Cc, Nn,
      0, (size_t)Nn * Cc, (size_t)Cc * Nn, 0,
      nullptr, yb, nullptr, nullptr, chsum, chsq);

  // --- BN finalize + residual relu ---
  k_bnstat<<<(Cc + 255) / 256, 256, 0, stream>>>(chsum, chsq, meanv, rstdv);
  k_final<<<(Bb * Cc * Nn) / 256, 256, 0, stream>>>(yb, x, meanv, rstdv, bnw, bnb, out);
}
